// COMETGate_15745350107663
// MI455X (gfx1250) — matrix operand order repaired and checked by reading, not compiled, not run on hardware
//
#include <hip/hip_runtime.h>

#define B_ROWS 8192
#define DIN    1024
#define DOUT   1024
#define NE     16
#define KK     8
#define NCOLP  256   // 248 real columns padded to 256

typedef __attribute__((ext_vector_type(16))) __bf16 bf16x16;
typedef __attribute__((ext_vector_type(8)))  __bf16 bf16x8;
typedef __attribute__((ext_vector_type(8)))  float  f32x8;
typedef __attribute__((ext_vector_type(4)))  float  f32x4;

__device__ __forceinline__ __bf16 f2bf(float f) {
  unsigned u = __builtin_bit_cast(unsigned, f);
  u += 0x7FFFu + ((u >> 16) & 1u);          // round-to-nearest-even
  unsigned short h = (unsigned short)(u >> 16);
  return __builtin_bit_cast(__bf16, h);
}

// ---------------------------------------------------------------------------
// Prep B: pack [Wz | Ww] (fp32, ndk layout) into the CDNA5 bf16 B-operand
// layout for V_WMMA_F32_16X16X32_BF16:
//   group (kc, ct): lane L holds column c = ct*16 + (L&15),
//   K values = (L>>4)*8 + (idx&7) + (idx>>3)*16 (same order as the A pack).
// Also fuse biases into a 256-entry vector (0 on pad columns).
// ---------------------------------------------------------------------------
__global__ void prep_w_kernel(const float* __restrict__ Wz, const float* __restrict__ bz,
                              const float* __restrict__ Ww, const float* __restrict__ bw,
                              __bf16* __restrict__ Bpack, float* __restrict__ biasPack) {
  int tid  = blockIdx.x * blockDim.x + threadIdx.x;   // 32*16*32*16 = 262144
  int idx  = tid & 15;
  int lane = (tid >> 4) & 31;
  int ct   = (tid >> 9) & 15;
  int kc   = tid >> 13;
  int c = ct * 16 + (lane & 15);
  int d = kc * 32 + (lane >> 4) * 8 + (idx & 7) + (idx >> 3) * 16;
  float v = 0.f;
  if (c < 120)      v = Wz[((c >> 3) * DIN + d) * KK + (c & 7)];
  else if (c < 248) { int cc = c - 120; v = Ww[((cc >> 3) * DIN + d) * KK + (cc & 7)]; }
  Bpack[tid] = f2bf(v);
  if (tid < NCOLP)
    biasPack[tid] = (tid < 120) ? bz[tid] : ((tid < 248) ? bw[tid - 120] : 0.f);
}

// ---------------------------------------------------------------------------
// Prep A: pack x (fp32 row-major) into the 16-bit WMMA A-operand layout so the
// GEMM hot loop does a single contiguous 32B load per lane per K-chunk:
//   lane (r = L&15, h = L>>4), elements 0..7  -> d = kc*32 + h*8 + i
//                               elements 8..15 -> d = kc*32 + 16 + h*8 + i
// ---------------------------------------------------------------------------
__global__ void prep_x_kernel(const float* __restrict__ x, __bf16* __restrict__ xpack) {
  int tid  = blockIdx.x * blockDim.x + threadIdx.x;   // 2*32*32*512 = 1048576
  int ge   = tid & 1;           // which 8-element group
  int lane = (tid >> 1) & 31;
  int kc   = (tid >> 6) & 31;
  int rt   = tid >> 11;
  int row  = rt * 16 + (lane & 15);
  int d0   = kc * 32 + (lane >> 4) * 8 + ge * 16;
  const f32x4* xp = (const f32x4*)(x + (size_t)row * DIN + d0);
  f32x4 a0 = xp[0], a1 = xp[1];
  bf16x8 o;
#pragma unroll
  for (int i = 0; i < 4; ++i) { o[i] = f2bf(a0[i]); o[i + 4] = f2bf(a1[i]); }
  *(bf16x8*)(xpack + ((size_t)(rt * 32 + kc) * 32 + lane) * 16 + ge * 8) = o;
}

// ---------------------------------------------------------------------------
// GEMM: scores[b, c] = sum_d x[b,d] * W[d,c]   (8192x1024 @ 1024x256)
// One wave = 16 rows x 64 cols (4 accumulators of 16x16), K swept in 32-chunks.
// Low register footprint (~80 VGPRs) -> high occupancy hides L2 latency;
// sched_group_barrier pins each K-chunk to: 10 VMEM reads, then 4 WMMAs.
// ---------------------------------------------------------------------------
__global__ void __launch_bounds__(256) gemm_kernel(const __bf16* __restrict__ xpack,
                                                   const __bf16* __restrict__ Bpack,
                                                   float* __restrict__ scores) {
  const int lane = threadIdx.x & 31;
  const int wid  = blockIdx.x * 8 + (threadIdx.x >> 5);
  const int rt   = wid >> 2;          // 16-row tile index (0..511)
  const int qt   = wid & 3;           // which 64-column quarter
  const int r    = lane & 15;
  const int h    = lane >> 4;

  f32x8 acc[4] = {};

  const __bf16* abase  = xpack + ((size_t)rt * 32 * 32 + lane) * 16;
  const __bf16* bbase0 = Bpack + ((size_t)(qt * 4) * 32 + lane) * 16;

  for (int kc = 0; kc < 32; ++kc) {
    bf16x16 af = *(const bf16x16*)(abase + (size_t)kc * 32 * 16);
    const __bf16* bbase = bbase0 + (size_t)kc * 16 * 32 * 16;
    bf16x16 bfrag[4];
#pragma unroll
    for (int t = 0; t < 4; ++t)
      bfrag[t] = *(const bf16x16*)(bbase + (size_t)t * 32 * 16);
#pragma unroll
    for (int t = 0; t < 4; ++t)
      acc[t] = __builtin_amdgcn_wmma_f32_16x16x32_bf16(
          false, af, false, bfrag[t], (short)0, acc[t], false, false);
#if __has_builtin(__builtin_amdgcn_sched_group_barrier)
    __builtin_amdgcn_sched_group_barrier(0x020, 10, 0);  // 10 VMEM reads (A + 4xB)
    __builtin_amdgcn_sched_group_barrier(0x008, 4, 0);   // 4 WMMA back-to-back
#endif
  }

  // C/D layout: VGPR rr -> M = rr + h*8, lane&15 -> N
#pragma unroll
  for (int t = 0; t < 4; ++t) {
    const int col = (qt * 4 + t) * 16 + r;
#pragma unroll
    for (int rr = 0; rr < 8; ++rr) {
      int row = rt * 16 + rr + h * 8;
      scores[(size_t)row * NCOLP + col] = acc[t][rr];
    }
  }
}

// ---------------------------------------------------------------------------
// Gate: per-row smoothstep -> 4-level tree probabilities -> stable exp
// (softmax denominator cancels against final normalization) -> permutation
// mix -> normalized 16-expert gate.
// ---------------------------------------------------------------------------
__global__ void gate_kernel(const float* __restrict__ scores,
                            const float* __restrict__ biasPack,
                            const float* __restrict__ P,
                            float* __restrict__ gate) {
  int b = blockIdx.x * blockDim.x + threadIdx.x;
  if (b >= B_ROWS) return;
  const float* row = scores + (size_t)b * NCOLP;

  float sc[KK][NE];
  float mx = -3.4028234663852886e38f;
#pragma unroll
  for (int k = 0; k < KK; ++k) {
    float ss[15];
#pragma unroll
    for (int n = 0; n < 15; ++n) {
      float v = row[n * 8 + k] + biasPack[n * 8 + k];
      float poly = -2.f * v * v * v + 1.5f * v + 0.5f;
      ss[n] = (v <= -0.5f) ? 0.f : ((v >= 0.5f) ? 1.f : poly);
    }
#pragma unroll
    for (int j = 0; j < NE; ++j) {
      float s0 = ss[0];               float p0 = ((j >> 3) & 1) ? 1.f - s0 : s0;
      float s1 = ss[1 + (j >> 3)];    float p1 = ((j >> 2) & 1) ? 1.f - s1 : s1;
      float s2 = ss[3 + (j >> 2)];    float p2 = ((j >> 1) & 1) ? 1.f - s2 : s2;
      float s3 = ss[7 + (j >> 1)];    float p3 = (j & 1)        ? 1.f - s3 : s3;
      float pr = p0 * p1 * p2 * p3;
      float lp = (pr <= 0.f) ? -3.4028234663852886e38f : __logf(pr + 1e-8f);
      float a  = row[120 + j * 8 + k] + biasPack[120 + j * 8 + k];
      float s  = a + lp;
      sc[k][j] = s;
      mx = fmaxf(mx, s);
    }
  }
#pragma unroll
  for (int k = 0; k < KK; ++k)
#pragma unroll
    for (int j = 0; j < NE; ++j)
      sc[k][j] = __expf(sc[k][j] - mx);   // un-normalized softmax (scale cancels)

  float g[NE];
  float gs = 0.f;
  for (int l = 0; l < NE; ++l) {
    float t = 0.f;
    for (int k = 0; k < KK; ++k)
#pragma unroll
      for (int j = 0; j < NE; ++j)
        t += sc[k][j] * P[(k * NE + j) * NE + l];
    g[l] = t;
    gs += t;
  }
  float ginv = 1.f / gs;
#pragma unroll
  for (int l = 0; l < NE; ++l) gate[b * NE + l] = g[l] * ginv;
}

// ---------------------------------------------------------------------------
// Output: y[b,d] = sum_n f[b,d,n] * gate[b,n]. 512 MB read-once stream of f:
// nontemporal b128 loads, one block per batch row, gate via LDS broadcast.
// ---------------------------------------------------------------------------
__global__ void __launch_bounds__(256) out_kernel(const float* __restrict__ f,
                                                  const float* __restrict__ gate,
                                                  float* __restrict__ y) {
  __shared__ float gsh[NE];
  const int b = blockIdx.x;
  if (threadIdx.x < NE) gsh[threadIdx.x] = gate[b * NE + threadIdx.x];
  __syncthreads();
  float gr[NE];
#pragma unroll
  for (int e = 0; e < NE; ++e) gr[e] = gsh[e];

  const float* fb = f + (size_t)b * DOUT * NE;
#pragma unroll
  for (int i = 0; i < 4; ++i) {
    const int d = threadIdx.x + i * 256;
    const f32x4* fp = (const f32x4*)(fb + (size_t)d * NE);
    __builtin_prefetch((const void*)(fp + 1024), 0, 1);   // 16 KB ahead in the stream
    f32x4 v0 = __builtin_nontemporal_load(fp);
    f32x4 v1 = __builtin_nontemporal_load(fp + 1);
    f32x4 v2 = __builtin_nontemporal_load(fp + 2);
    f32x4 v3 = __builtin_nontemporal_load(fp + 3);
    float acc = 0.f;
#pragma unroll
    for (int e = 0; e < 4; ++e)
      acc += v0[e] * gr[e] + v1[e] * gr[4 + e] + v2[e] * gr[8 + e] + v3[e] * gr[12 + e];
    y[(size_t)b * DOUT + d] = acc;
  }
}

// ---------------------------------------------------------------------------
extern "C" void kernel_launch(void* const* d_in, const int* in_sizes, int n_in,
                              void* d_out, int out_size, void* d_ws, size_t ws_size,
                              hipStream_t stream) {
  const float* f  = (const float*)d_in[0];   // (8192, 1024, 16)
  const float* x  = (const float*)d_in[1];   // (8192, 1024)
  const float* P  = (const float*)d_in[2];   // (8, 16, 16)
  const float* Wz = (const float*)d_in[3];   // (15, 1024, 8)
  const float* bz = (const float*)d_in[4];   // (15, 8)
  const float* Ww = (const float*)d_in[5];   // (16, 1024, 8)
  const float* bw = (const float*)d_in[6];   // (16, 8)
  float* y = (float*)d_out;

  char* ws = (char*)d_ws;
  __bf16* Bpack    = (__bf16*)ws;                               // 512 KB
  float*  biasPack = (float*)(ws + 524288);                     // 4 KB slot
  float*  gate     = (float*)(ws + 528384);                     // 512 KB
  float*  scores   = (float*)(ws + 1052672);                    // 8 MB
  __bf16* xpack    = (__bf16*)(ws + 9441280);                   // 16 MB

  prep_w_kernel<<<1024, 256, 0, stream>>>(Wz, bz, Ww, bw, Bpack, biasPack);
  prep_x_kernel<<<4096, 256, 0, stream>>>(x, xpack);
  gemm_kernel<<<256, 256, 0, stream>>>(xpack, Bpack, scores);   // 2048 waves
  gate_kernel<<<32, 256, 0, stream>>>(scores, biasPack, P, gate);
  out_kernel<<<B_ROWS, 256, 0, stream>>>(f, gate, y);
}
